// BertFlashFWSVDBlock_85529978733267
// MI455X (gfx1250) — compile-verified, matrix operand order checked by reading, not run on hardware
//
#include <hip/hip_runtime.h>
#include <hip/hip_bf16.h>
#include <math.h>

// ---------------- types ----------------
typedef __bf16 bf16_t;
typedef __attribute__((ext_vector_type(16))) __bf16 v16bf;
typedef __attribute__((ext_vector_type(8)))  float  v8f;

// ---------------- problem dims ----------------
#define BB   8
#define MM   512
#define DMOD 1024
#define HH   16
#define DHD  64
#define RATT 32
#define RWO  512
#define RFF  256
#define DFFD 4096
#define ROWS (BB * MM) /* 4096 */

// ---------------------------------------------------------------------------
// WMMA fragment helpers (CDNA5 wave32, 16x16x32 bf16 -> f32)
// A-fragment (16 rows x 32 K): lane 0-15 -> row=lane, Kbase 0; lane 16-31 ->
// row=lane-16, Kbase 8. VGPR j packs K = kbase + (j>>2)*16 + (j&3)*2, +1.
// => per lane two contiguous 16B chunks at (row*ld + kb) and (+16 elements).
// B-fragment = same layout applied to B^T, so B tiles are staged N-major.
// C/D: VGPR r holds (M = r + 8*(lane>=16), N = lane%16).
// ---------------------------------------------------------------------------
__device__ __forceinline__ v16bf load_frag16x32(const bf16_t* p, int ld) {
  int lane = threadIdx.x & 31;
  int r    = lane & 15;
  int kb   = (lane >> 4) << 3;
  const bf16_t* q = p + r * ld + kb;
  v16bf f;
  ((uint4*)&f)[0] = *(const uint4*)(q);
  ((uint4*)&f)[1] = *(const uint4*)(q + 16);
  return f;
}

__device__ __forceinline__ v8f wmma_bf16f32(v16bf a, v16bf b, v8f c) {
  return __builtin_amdgcn_wmma_f32_16x16x32_bf16(
      false, a, false, b, (short)0, c, false, false);
}

// ---------------------------------------------------------------------------
// 16-lane butterfly reductions, pure VALU via DPP16 (no LDS traffic).
// 0xB1 = quad_perm[1,0,3,2] (xor1), 0x4E = quad_perm[2,3,0,1] (xor2),
// 0x141 = row_half_mirror (xor4-equivalent), 0x140 = row_mirror (xor8-equiv).
// ---------------------------------------------------------------------------
__device__ __forceinline__ float wave16_max(float x) {
  float t;
  t = __int_as_float(__builtin_amdgcn_mov_dpp(__float_as_int(x), 0xB1, 0xF, 0xF, true));
  x = fmaxf(x, t);
  t = __int_as_float(__builtin_amdgcn_mov_dpp(__float_as_int(x), 0x4E, 0xF, 0xF, true));
  x = fmaxf(x, t);
  t = __int_as_float(__builtin_amdgcn_mov_dpp(__float_as_int(x), 0x141, 0xF, 0xF, true));
  x = fmaxf(x, t);
  t = __int_as_float(__builtin_amdgcn_mov_dpp(__float_as_int(x), 0x140, 0xF, 0xF, true));
  x = fmaxf(x, t);
  return x;
}
__device__ __forceinline__ float wave16_sum(float x) {
  float t;
  t = __int_as_float(__builtin_amdgcn_mov_dpp(__float_as_int(x), 0xB1, 0xF, 0xF, true));
  x += t;
  t = __int_as_float(__builtin_amdgcn_mov_dpp(__float_as_int(x), 0x4E, 0xF, 0xF, true));
  x += t;
  t = __int_as_float(__builtin_amdgcn_mov_dpp(__float_as_int(x), 0x141, 0xF, 0xF, true));
  x += t;
  t = __int_as_float(__builtin_amdgcn_mov_dpp(__float_as_int(x), 0x140, 0xF, 0xF, true));
  x += t;
  return x;
}

// Async global->LDS 16B copy (CDNA5, ASYNCcnt-tracked; bypasses VGPRs).
__device__ __forceinline__ void async_copy_b128(unsigned lds_off,
                                                unsigned long long gaddr) {
  asm volatile("global_load_async_to_lds_b128 %0, %1, off"
               :: "v"(lds_off), "v"(gaddr) : "memory");
}
__device__ __forceinline__ void async_wait0() {
  asm volatile("s_wait_asynccnt 0" ::: "memory");
}

// ---------------------------------------------------------------------------
// Generic batched tiled GEMM:  C[z] = act(A[z] @ B[z] + bias[z] (+ resid))
// grid = (N/BN, M/BM, numZ), block = 256 (8 waves), K % 32 == 0.
// Offsets per z: off = (z/inner)*strideOuter + (z%inner)*strideInner.
// ---------------------------------------------------------------------------
template <int BM, int BN, int WM, int WN, bool ABF16, bool ASYNCA, bool BIAS,
          bool RESID, bool GELU, bool OUTBF16>
__global__ __launch_bounds__(256) void gemm_kernel(
    const void* __restrict__ Aptr, const float* __restrict__ Bptr,
    const float* __restrict__ biasPtr, const float* __restrict__ residPtr,
    void* __restrict__ Cptr, int Mdim, int Ndim, int Kdim, int inner,
    long aO, long aI, long bO, long bI, long cO, long cI, long biasI) {
  (void)Mdim;
  static_assert((BM / WM) * (BN / WN) == 8, "8 waves");
  int z = blockIdx.z;
  long zo = z / inner, zi = z % inner;
  long aoff = zo * aO + zi * aI;
  long coff = zo * cO + zi * cI;
  const float* Bb = Bptr + zo * bO + zi * bI;

  __shared__ __align__(16) bf16_t sA[BM][40];  // +8 pad keeps 16B alignment
  __shared__ __align__(16) bf16_t sB[BN][40];  // B staged transposed: [n][k]

  int m0 = blockIdx.y * BM, n0 = blockIdx.x * BN;
  constexpr int WX = BN / WN;
  int wave = threadIdx.x >> 5;
  int wy = wave / WX, wx = wave % WX;
  int lane = threadIdx.x & 31, hi = lane >> 4, nc = lane & 15;

  v8f acc[WM / 16][WN / 16] = {};

  for (int k0 = 0; k0 < Kdim; k0 += 32) {
    if (ASYNCA) {
      // A already bf16 in global: contiguous 16B row chunks -> async to LDS.
      for (int i = threadIdx.x; i < BM * 4; i += 256) {
        int r = i >> 2, c = (i & 3) * 8;
        unsigned lds = (unsigned)(uintptr_t)&sA[r][c];
        unsigned long long g = (unsigned long long)(uintptr_t)(
            (const bf16_t*)Aptr + aoff + (long)(m0 + r) * Kdim + k0 + c);
        async_copy_b128(lds, g);
      }
      async_wait0();
    } else {
      for (int i = threadIdx.x; i < BM * 32; i += 256) {
        int r = i >> 5, c = i & 31;
        long ai = aoff + (long)(m0 + r) * Kdim + k0 + c;
        float va = ABF16 ? (float)((const bf16_t*)Aptr)[ai]
                         : ((const float*)Aptr)[ai];
        sA[r][c] = (bf16_t)va;
      }
    }
    for (int i = threadIdx.x; i < BN * 32; i += 256) {
      int r = i >> 5, c = i & 31;  // r = n, c = k
      sB[r][c] = (bf16_t)Bb[(long)(k0 + c) * Ndim + n0 + r];
    }
    // Prefetch next k-tile (lowers to global_prefetch_b8).
    if (k0 + 32 < Kdim) {
      int t = threadIdx.x;
      if (t < BM) {
        if (ABF16)
          __builtin_prefetch((const bf16_t*)Aptr + aoff +
                                 (long)(m0 + t) * Kdim + k0 + 32, 0, 0);
        else
          __builtin_prefetch((const float*)Aptr + aoff +
                                 (long)(m0 + t) * Kdim + k0 + 32, 0, 0);
      }
      if (t >= 224) __builtin_prefetch(Bb + (long)(k0 + 32 + t - 224) * Ndim + n0, 0, 0);
    }
    __syncthreads();
#pragma unroll
    for (int mi = 0; mi < WM / 16; ++mi) {
      v16bf af = load_frag16x32(&sA[wy * WM + mi * 16][0], 40);
#pragma unroll
      for (int ni = 0; ni < WN / 16; ++ni) {
        v16bf bf = load_frag16x32(&sB[wx * WN + ni * 16][0], 40);
        acc[mi][ni] = wmma_bf16f32(af, bf, acc[mi][ni]);
      }
    }
    __syncthreads();
  }

#pragma unroll
  for (int mi = 0; mi < WM / 16; ++mi)
#pragma unroll
    for (int ni = 0; ni < WN / 16; ++ni)
#pragma unroll
      for (int r = 0; r < 8; ++r) {
        int row = m0 + wy * WM + mi * 16 + hi * 8 + r;
        int col = n0 + wx * WN + ni * 16 + nc;
        float val = acc[mi][ni][r];
        if (BIAS)  val += biasPtr[zi * biasI + col];
        if (RESID) val += residPtr[(long)row * Ndim + col];
        if (GELU)  val = 0.5f * val * (1.0f + erff(val * 0.70710678118654752f));
        long off = coff + (long)row * Ndim + col;
        if (OUTBF16) ((bf16_t*)Cptr)[off] = (bf16_t)val;
        else         ((float*)Cptr)[off]  = val;
      }
}

// ---------------------------------------------------------------------------
// Flash attention: grid = (M/128, B*H), block = 256 (8 waves).
// Each wave owns 16 query rows; iterates 32-key tiles with online softmax.
// q/k/v are bf16 [B,H,M,DH]; output written to attn_out f32 [B,M,DM].
// ---------------------------------------------------------------------------
__global__ __launch_bounds__(256) void attn_kernel(
    const bf16_t* __restrict__ q, const bf16_t* __restrict__ k,
    const bf16_t* __restrict__ v, const float* __restrict__ mask,
    float* __restrict__ attn_out) {
  int bh = blockIdx.y;
  int b = bh >> 4, h = bh & 15;
  int wave = threadIdx.x >> 5, lane = threadIdx.x & 31;
  int hi = lane >> 4, nc = lane & 15;
  int m0 = blockIdx.x * 128 + wave * 16;

  const bf16_t* qb = q + (long)bh * MM * DHD;
  const bf16_t* kb = k + (long)bh * MM * DHD;
  const bf16_t* vb = v + (long)bh * MM * DHD;

  __shared__ __align__(16) bf16_t sK[32][64];      // [key][dh]
  __shared__ __align__(16) bf16_t sV[64][32];      // transposed: [dh][key]
  __shared__ __align__(16) bf16_t sP[8][16][32];   // per-wave P staging

  v16bf qf0 = load_frag16x32(qb + (long)m0 * DHD, DHD);       // K dims 0..31
  v16bf qf1 = load_frag16x32(qb + (long)m0 * DHD + 32, DHD);  // K dims 32..63

  v8f o0 = {}, o1 = {}, o2 = {}, o3 = {};
  float rmax[8], rsum[8];
#pragma unroll
  for (int r = 0; r < 8; ++r) { rmax[r] = -1e30f; rsum[r] = 0.f; }

  for (int kt = 0; kt < MM / 32; ++kt) {
    // sK: contiguous 4KB global->LDS copy via async b128 (16B per thread).
    {
      int t = threadIdx.x;
      unsigned lds = (unsigned)(uintptr_t)&sK[0][0] + (unsigned)t * 16u;
      unsigned long long g = (unsigned long long)(uintptr_t)(
          kb + (long)kt * 32 * DHD + t * 8);
      async_copy_b128(lds, g);
    }
    // sV: transposed staging (element-wise).
    for (int i = threadIdx.x; i < 32 * 64; i += 256) {
      int kr = i >> 6, c = i & 63;
      sV[c][kr] = vb[(long)(kt * 32 + kr) * DHD + c];
    }
    async_wait0();
    __syncthreads();

    v8f s0 = {}, s1 = {};
    s0 = wmma_bf16f32(qf0, load_frag16x32(&sK[0][0], 64), s0);
    s0 = wmma_bf16f32(qf1, load_frag16x32(&sK[0][0] + 32, 64), s0);
    s1 = wmma_bf16f32(qf0, load_frag16x32(&sK[16][0], 64), s1);
    s1 = wmma_bf16f32(qf1, load_frag16x32(&sK[16][0] + 32, 64), s1);

    float mk0 = mask[b * MM + kt * 32 + nc];
    float mk1 = mask[b * MM + kt * 32 + 16 + nc];

#pragma unroll
    for (int r = 0; r < 8; ++r) {
      float a0 = s0[r] * 0.125f + mk0;  // 1/sqrt(64)
      float a1 = s1[r] * 0.125f + mk1;
      float mx = wave16_max(fmaxf(a0, a1));
      float nm = fmaxf(rmax[r], mx);
      float corr = __expf(rmax[r] - nm);
      rmax[r] = nm;
      float p0 = __expf(a0 - nm);
      float p1 = __expf(a1 - nm);
      float rs = wave16_sum(p0 + p1);
      rsum[r] = rsum[r] * corr + rs;
      o0[r] *= corr; o1[r] *= corr; o2[r] *= corr; o3[r] *= corr;
      sP[wave][hi * 8 + r][nc]      = (bf16_t)p0;
      sP[wave][hi * 8 + r][16 + nc] = (bf16_t)p1;
    }
    // same-wave DS store -> DS read: wait LDS counter, fence compiler reorder
    asm volatile("s_wait_dscnt 0" ::: "memory");
    v16bf pf = load_frag16x32(&sP[wave][0][0], 32);
    o0 = wmma_bf16f32(pf, load_frag16x32(&sV[0][0],  32), o0);
    o1 = wmma_bf16f32(pf, load_frag16x32(&sV[16][0], 32), o1);
    o2 = wmma_bf16f32(pf, load_frag16x32(&sV[32][0], 32), o2);
    o3 = wmma_bf16f32(pf, load_frag16x32(&sV[48][0], 32), o3);
    __syncthreads();
  }

#pragma unroll
  for (int r = 0; r < 8; ++r) {
    float inv = 1.0f / rsum[r];
    float* row = attn_out + ((long)b * MM + m0 + hi * 8 + r) * DMOD +
                 h * DHD + nc;
    row[0]  = o0[r] * inv;
    row[16] = o1[r] * inv;
    row[32] = o2[r] * inv;
    row[48] = o3[r] * inv;
  }
}

// ---------------------------------------------------------------------------
// LayerNorm over last dim (1024), one block per row. eps = 1e-12 (BERT).
// ---------------------------------------------------------------------------
__global__ __launch_bounds__(256) void ln_kernel(
    const float* __restrict__ in, const float* __restrict__ w,
    const float* __restrict__ bb, float* __restrict__ out) {
  int row = blockIdx.x, t = threadIdx.x;
  const float* p = in + (long)row * DMOD;
  float v[4]; float s = 0.f, ss = 0.f;
#pragma unroll
  for (int i = 0; i < 4; ++i) {
    v[i] = p[t + i * 256];
    s += v[i]; ss += v[i] * v[i];
  }
  __shared__ float rs[256], rq[256];
  rs[t] = s; rq[t] = ss;
  __syncthreads();
  for (int off = 128; off > 0; off >>= 1) {
    if (t < off) { rs[t] += rs[t + off]; rq[t] += rq[t + off]; }
    __syncthreads();
  }
  float mu  = rs[0] * (1.0f / DMOD);
  float var = rq[0] * (1.0f / DMOD) - mu * mu;
  float rinv = rsqrtf(var + 1e-12f);
  float* o = out + (long)row * DMOD;
#pragma unroll
  for (int i = 0; i < 4; ++i) {
    int c = t + i * 256;
    o[c] = (v[i] - mu) * rinv * w[c] + bb[c];
  }
}

// ---------------------------------------------------------------------------
// Host orchestration. Workspace aliased; peak 80 MiB (see round-1 map).
// ---------------------------------------------------------------------------
extern "C" void kernel_launch(void* const* d_in, const int* in_sizes, int n_in,
                              void* d_out, int out_size, void* d_ws,
                              size_t ws_size, hipStream_t stream) {
  (void)in_sizes; (void)n_in; (void)out_size; (void)ws_size;

  const float* x    = (const float*)d_in[0];
  const float* mask = (const float*)d_in[1];
  const float* Pq   = (const float*)d_in[2];
  const float* Vq   = (const float*)d_in[3];
  const float* bq   = (const float*)d_in[4];
  const float* Pk   = (const float*)d_in[5];
  const float* Vk   = (const float*)d_in[6];
  const float* bk   = (const float*)d_in[7];
  const float* Pv   = (const float*)d_in[8];
  const float* Vv   = (const float*)d_in[9];
  const float* bv   = (const float*)d_in[10];
  const float* Uo   = (const float*)d_in[11];
  const float* Vo   = (const float*)d_in[12];
  const float* bo   = (const float*)d_in[13];
  const float* U1   = (const float*)d_in[14];
  const float* V1   = (const float*)d_in[15];
  const float* b1   = (const float*)d_in[16];
  const float* U2   = (const float*)d_in[17];
  const float* V2   = (const float*)d_in[18];
  const float* b2   = (const float*)d_in[19];
  const float* ln1w = (const float*)d_in[20];
  const float* ln1b = (const float*)d_in[21];
  const float* ln2w = (const float*)d_in[22];
  const float* ln2b = (const float*)d_in[23];

  char* ws = (char*)d_ws;
  const size_t MB1 = 1ull << 20;
  float*  lowq  = (float*)(ws + 0);
  float*  lowk  = (float*)(ws + 8 * MB1);
  float*  lowv  = (float*)(ws + 16 * MB1);
  bf16_t* qb    = (bf16_t*)(ws + 24 * MB1);
  bf16_t* kbp   = (bf16_t*)(ws + 32 * MB1);
  bf16_t* vbp   = (bf16_t*)(ws + 40 * MB1);
  float*  attnO = (float*)(ws + 48 * MB1);
  float*  x1    = (float*)(ws + 64 * MB1);
  float*  t1    = (float*)(ws + 0);
  float*  f1    = (float*)(ws + 8 * MB1);
  bf16_t* hb    = (bf16_t*)(ws + 12 * MB1);
  float*  f2    = (float*)(ws + 44 * MB1);
  float*  t2    = (float*)(ws + 48 * MB1);

  dim3 blk(256);
  const long zero = 0;

  // ---- stage 1: low = x @ P   per (b,h): [512,1024]x[1024,32] ----
  {
    long aO = (long)MM * DMOD, bI = (long)DMOD * RATT;
    long cO = (long)HH * MM * RATT, cI = (long)MM * RATT;
    dim3 g(1, MM / 128, BB * HH);
    gemm_kernel<128, 32, 16, 32, false, false, false, false, false, false>
        <<<g, blk, 0, stream>>>(x, Pq, nullptr, nullptr, lowq, MM, RATT, DMOD,
                                HH, aO, zero, zero, bI, cO, cI, zero);
    gemm_kernel<128, 32, 16, 32, false, false, false, false, false, false>
        <<<g, blk, 0, stream>>>(x, Pk, nullptr, nullptr, lowk, MM, RATT, DMOD,
                                HH, aO, zero, zero, bI, cO, cI, zero);
    gemm_kernel<128, 32, 16, 32, false, false, false, false, false, false>
        <<<g, blk, 0, stream>>>(x, Pv, nullptr, nullptr, lowv, MM, RATT, DMOD,
                                HH, aO, zero, zero, bI, cO, cI, zero);
  }
  // ---- stage 2: q/k/v = low @ V + b  -> bf16 [B,H,M,64] ----
  {
    long aO = (long)HH * MM * RATT, aI = (long)MM * RATT;
    long bI = (long)RATT * DHD;
    long cO = (long)HH * MM * DHD, cI = (long)MM * DHD;
    dim3 g(1, MM / 128, BB * HH);
    gemm_kernel<128, 64, 32, 32, false, false, true, false, false, true>
        <<<g, blk, 0, stream>>>(lowq, Vq, bq, nullptr, qb, MM, DHD, RATT, HH,
                                aO, aI, zero, bI, cO, cI, (long)DHD);
    gemm_kernel<128, 64, 32, 32, false, false, true, false, false, true>
        <<<g, blk, 0, stream>>>(lowk, Vk, bk, nullptr, kbp, MM, DHD, RATT, HH,
                                aO, aI, zero, bI, cO, cI, (long)DHD);
    gemm_kernel<128, 64, 32, 32, false, false, true, false, false, true>
        <<<g, blk, 0, stream>>>(lowv, Vv, bv, nullptr, vbp, MM, DHD, RATT, HH,
                                aO, aI, zero, bI, cO, cI, (long)DHD);
  }
  // ---- stage 3: flash attention -> attn_out f32 [B,M,DM] ----
  attn_kernel<<<dim3(MM / 128, BB * HH), blk, 0, stream>>>(qb, kbp, vbp, mask,
                                                           attnO);
  // ---- stage 4: t1 = attn_out @ Uo ; t2 = t1 @ Vo + bo + x ; LN1 -> x1 ----
  gemm_kernel<128, 128, 32, 64, false, false, false, false, false, false>
      <<<dim3(RWO / 128, ROWS / 128, 1), blk, 0, stream>>>(
          attnO, Uo, nullptr, nullptr, t1, ROWS, RWO, DMOD, 1, zero, zero,
          zero, zero, zero, zero, zero);
  gemm_kernel<128, 128, 32, 64, false, false, true, true, false, false>
      <<<dim3(DMOD / 128, ROWS / 128, 1), blk, 0, stream>>>(
          t1, Vo, bo, x, t2, ROWS, DMOD, RWO, 1, zero, zero, zero, zero, zero,
          zero, zero);
  ln_kernel<<<ROWS, blk, 0, stream>>>(t2, ln1w, ln1b, x1);
  // ---- stage 5: FFN ----
  gemm_kernel<128, 128, 32, 64, false, false, false, false, false, false>
      <<<dim3(RFF / 128, ROWS / 128, 1), blk, 0, stream>>>(
          x1, U1, nullptr, nullptr, f1, ROWS, RFF, DMOD, 1, zero, zero, zero,
          zero, zero, zero, zero);
  gemm_kernel<128, 128, 32, 64, false, false, true, false, true, true>
      <<<dim3(DFFD / 128, ROWS / 128, 1), blk, 0, stream>>>(
          f1, V1, b1, nullptr, hb, ROWS, DFFD, RFF, 1, zero, zero, zero, zero,
          zero, zero, zero);
  gemm_kernel<128, 128, 32, 64, true, true, false, false, false, false>
      <<<dim3(RFF / 128, ROWS / 128, 1), blk, 0, stream>>>(
          hb, U2, nullptr, nullptr, f2, ROWS, RFF, DFFD, 1, zero, zero, zero,
          zero, zero, zero, zero);
  gemm_kernel<128, 128, 32, 64, false, false, true, true, false, false>
      <<<dim3(DMOD / 128, ROWS / 128, 1), blk, 0, stream>>>(
          f2, V2, b2, x1, t2, ROWS, DMOD, RFF, 1, zero, zero, zero, zero, zero,
          zero, zero);
  // ---- stage 6: LN2 -> output ----
  ln_kernel<<<ROWS, blk, 0, stream>>>(t2, ln2w, ln2b, (float*)d_out);
}
